// QuantumLayer_13709535609634
// MI455X (gfx1250) — matrix operand order repaired and checked by reading, not compiled, not run on hardware
//
#include <hip/hip_runtime.h>
#include <hip/hip_fp16.h>

typedef __attribute__((ext_vector_type(16))) _Float16 v16h;
typedef __attribute__((ext_vector_type(8)))  float    v8f;
typedef __attribute__((ext_vector_type(2)))  float    v2f;

#define TOKENS (8 * 2048)

// ---------------------------------------------------------------------------
// Kernel 1: build phase-folded real matrices Are/Aim per split such that
//   phi_re = Are @ R,  phi_im = Aim @ R,  R[idx][token] = prod(cos/sin)
// where column idx of the circuit unitary U absorbs the embedding phase
// (-i)^popcount(idx). Qubit q <-> bit (3-q), mask(q) = 8 >> q.
// ---------------------------------------------------------------------------
template <int MK>
__device__ __forceinline__ void rot_xyz(float2 st[16], float w) {
  float ch = __cosf(0.5f * w), sh = __sinf(0.5f * w);
  // RX
#pragma unroll
  for (int i = 0; i < 16; ++i)
    if (!(i & MK)) {
      float2 a = st[i], b = st[i | MK];
      st[i]      = make_float2(ch * a.x + sh * b.y, ch * a.y - sh * b.x);
      st[i | MK] = make_float2(sh * a.y + ch * b.x, -sh * a.x + ch * b.y);
    }
  // RY
#pragma unroll
  for (int i = 0; i < 16; ++i)
    if (!(i & MK)) {
      float2 a = st[i], b = st[i | MK];
      st[i]      = make_float2(ch * a.x - sh * b.x, ch * a.y - sh * b.y);
      st[i | MK] = make_float2(sh * a.x + ch * b.x, sh * a.y + ch * b.y);
    }
  // RZ
#pragma unroll
  for (int i = 0; i < 16; ++i) {
    float2 a = st[i];
    if (i & MK) st[i] = make_float2(ch * a.x - sh * a.y, ch * a.y + sh * a.x);
    else        st[i] = make_float2(ch * a.x + sh * a.y, ch * a.y - sh * a.x);
  }
}

template <int MC, int MT>
__device__ __forceinline__ void cnotg(float2 st[16]) {
#pragma unroll
  for (int i = 0; i < 16; ++i)
    if ((i & MC) && !(i & MT)) {
      float2 t = st[i]; st[i] = st[i | MT]; st[i | MT] = t;
    }
}

template <int M1, int M2>
__device__ __forceinline__ void swapg(float2 st[16]) {
#pragma unroll
  for (int i = 0; i < 16; ++i)
    if ((i & M1) && !(i & M2)) {
      int j = (i ^ M1) | M2;
      float2 t = st[i]; st[i] = st[j]; st[j] = t;
    }
}

template <int MK>
__device__ __forceinline__ void hadg(float2 st[16]) {
  const float r = 0.70710678118654752440f;
#pragma unroll
  for (int i = 0; i < 16; ++i)
    if (!(i & MK)) {
      float2 a = st[i], b = st[i | MK];
      st[i]      = make_float2(r * (a.x + b.x), r * (a.y + b.y));
      st[i | MK] = make_float2(r * (a.x - b.x), r * (a.y - b.y));
    }
}

__global__ void build_unitaries(const float* __restrict__ qw,
                                float* __restrict__ A) {
  int s = blockIdx.x;   // split 0..15
  int c = threadIdx.x;  // basis column 0..15
  float2 st[16];
#pragma unroll
  for (int i = 0; i < 16; ++i) st[i] = make_float2(0.f, 0.f);
  st[c] = make_float2(1.f, 0.f);
#pragma unroll
  for (int l = 0; l < 2; ++l) {
    const float* wl = qw + s * 8 + l * 4;
    rot_xyz<8>(st, wl[0]);
    rot_xyz<4>(st, wl[1]);
    rot_xyz<2>(st, wl[2]);
    rot_xyz<1>(st, wl[3]);
    cnotg<8, 4>(st);  // CNOT(0,1)
    swapg<4, 2>(st);  // SWAP(1,2)
    cnotg<2, 1>(st);  // CNOT(2,3)
    hadg<8>(st);      // H(0)
    cnotg<8, 4>(st);  // CNOT(0,1)
    cnotg<8, 2>(st);  // CNOT(0,2)
    cnotg<4, 1>(st);  // CNOT(1,3)
  }
  int p = ((c & 1) + ((c >> 1) & 1) + ((c >> 2) & 1) + ((c >> 3) & 1)) & 3;
  float* Ar = A + s * 512 + c;        // Are[s][i][c]
  float* Ai = A + s * 512 + 256 + c;  // Aim[s][i][c]
#pragma unroll
  for (int i = 0; i < 16; ++i) {
    float ur = st[i].x, ui = st[i].y, ar, ai;
    if (p == 0)      { ar =  ur; ai =  ui; }
    else if (p == 1) { ar =  ui; ai = -ur; }
    else if (p == 2) { ar = -ur; ai = -ui; }
    else             { ar = -ui; ai =  ur; }
    Ar[i * 16] = ar;
    Ai[i * 16] = ai;
  }
}

// ---------------------------------------------------------------------------
// Kernel 1b: pre-convert MLP weights to f16 (loaded as WMMA-B v16h directly).
// ---------------------------------------------------------------------------
__global__ void convert_weights(const float* __restrict__ W1,
                                const float* __restrict__ W2,
                                _Float16* __restrict__ W1h,
                                _Float16* __restrict__ W2h) {
  int i = blockIdx.x * blockDim.x + threadIdx.x;
  if (i < 64 * 192) W1h[i] = (_Float16)W1[i];
  else if (i < 64 * 192 + 64 * 64) {
    int j = i - 64 * 192;
    W2h[j] = (_Float16)W2[j];
  }
}

// ---------------------------------------------------------------------------
// fp32 WMMA (V_WMMA_F32_16X16X4_F32), signature probe-verified last round.
// ---------------------------------------------------------------------------
__device__ __forceinline__ v8f wmma_f32_k4(v2f a, v2f b, v8f c) {
#if __has_builtin(__builtin_amdgcn_wmma_f32_16x16x4_f32)
  return __builtin_amdgcn_wmma_f32_16x16x4_f32(false, a, false, b, (short)0, c,
                                               false, false);
#else
  int lane = threadIdx.x & 31;
  int lg = lane >> 4, n = lane & 15;
#pragma unroll
  for (int j = 0; j < 8; ++j) {
    int mrow = j + 8 * lg;
    float acc = c[j];
#pragma unroll
    for (int k = 0; k < 4; ++k) {
      float av = __shfl(k < 2 ? a[k] : a[k - 2], mrow + 16 * (k >> 1));
      float bv = __shfl(k < 2 ? b[k] : b[k - 2], n + 16 * (k >> 1));
      acc = fmaf(av, bv, acc);
    }
    c[j] = acc;
  }
  return c;
#endif
}

// ---------------------------------------------------------------------------
// Kernel 2: fused quantum features (fp32 WMMA) + LayerNorm + MLP (f16 WMMA).
// 64 threads = 2 waves; each wave owns a 16-token tile.
// lane: m = lane&15 (token / row-col), lg = lane>>4 (half-wave group).
// ---------------------------------------------------------------------------
__global__ __launch_bounds__(64) void fused_forward(
    const float* __restrict__ x, const float* __restrict__ A,
    const float* __restrict__ ln_g, const float* __restrict__ ln_b,
    const _Float16* __restrict__ W1h, const float* __restrict__ b1,
    const _Float16* __restrict__ W2h, const float* __restrict__ b2,
    float* __restrict__ out) {
  // Are|Aim interleaved rows: [s][i*36 + {0..15 re, 16..31 im, pad 4}].
  // Even stride 36 -> aligned float2 (ds_load_b64) A-operand loads,
  // bank-conflict-free across both half-waves.
  __shared__ float a_lds[16][16 * 36];   // 36,864 B
  __shared__ float q_lds[2][16][194];    // 24,832 B

  const int tid  = threadIdx.x;
  const int w    = tid >> 5;
  const int lane = tid & 31;
  const int lg   = lane >> 4;
  const int m    = lane & 15;

  __builtin_prefetch(W1h, 0, 1);  // global_prefetch_b8 into GL2
  __builtin_prefetch(W2h, 0, 1);

  for (int t = tid; t < 16 * 512; t += 64) {
    int c = t & 15, i = (t >> 4) & 15, hf = (t >> 8) & 1, s = t >> 9;
    a_lds[s][i * 36 + hf * 16 + c] = A[t];
  }
  __syncthreads();

  const int tok0  = (blockIdx.x * 2 + w) * 16;
  const int token = tok0 + m;
  const float4* xp = (const float4*)(x + token * 64);
  const int m36 = m * 36;

  // ---- quantum features: splits in pairs via fp32 WMMA ----
  for (int sp = 0; sp < 16; sp += 2) {
    v8f dre0 = {}, dim0 = {}, dre1 = {}, dim1 = {};
#pragma unroll
    for (int half = 0; half < 2; ++half) {
      const int s = sp + half;
      float4 ang = xp[s];
      float cv0 = __cosf(0.5f * ang.x), sv0 = __sinf(0.5f * ang.x);
      float cv1 = __cosf(0.5f * ang.y), sv1 = __sinf(0.5f * ang.y);
      float cv2 = __cosf(0.5f * ang.z), sv2 = __sinf(0.5f * ang.z);
      float cv3 = __cosf(0.5f * ang.w), sv3 = __sinf(0.5f * ang.w);
      float f2 = lg ? sv2 : cv2;  // idx bit1 = lane-group
      const float* As = &a_lds[s][0];
#pragma unroll
      for (int kc = 0; kc < 4; ++kc) {
        const int kb = 4 * kc + 2 * lg;  // B entries: idx = kb, kb+1
        float f0 = (kc & 2) ? sv0 : cv0;
        float f1 = (kc & 1) ? sv1 : cv1;
        float common = f0 * f1 * f2;
        v2f b;  b[0] = common * cv3;  b[1] = common * sv3;
        v2f ar = *(const v2f*)(As + m36 + kb);        // Are row m, k=kb..kb+1
        v2f ai = *(const v2f*)(As + m36 + 16 + kb);   // Aim row m
        if (half == 0) {
          dre0 = wmma_f32_k4(ar, b, dre0);
          dim0 = wmma_f32_k4(ai, b, dim0);
        } else {
          dre1 = wmma_f32_k4(ar, b, dre1);
          dim1 = wmma_f32_k4(ai, b, dim1);
        }
      }
    }
    // Exchange halves: lg=0 lanes assemble split sp, lg=1 split sp+1.
    float plR[8], phR[8], plI[8], phI[8];
#pragma unroll
    for (int j = 0; j < 8; ++j) {
      float oR0 = __shfl_xor(dre0[j], 16);
      float oI0 = __shfl_xor(dim0[j], 16);
      float oR1 = __shfl_xor(dre1[j], 16);
      float oI1 = __shfl_xor(dim1[j], 16);
      plR[j] = lg ? oR1 : dre0[j];   // states 0..7
      phR[j] = lg ? dre1[j] : oR0;   // states 8..15
      plI[j] = lg ? oI1 : dim0[j];
      phI[j] = lg ? dim1[j] : oI0;
    }
#define PRE(i) ((i) < 8 ? plR[(i)] : phR[(i) - 8])
#define PIM(i) ((i) < 8 ? plI[(i)] : phI[(i) - 8])
    float n2[16];
#pragma unroll
    for (int i = 0; i < 16; ++i) n2[i] = PRE(i) * PRE(i) + PIM(i) * PIM(i);
    float* qrow = &q_lds[w][m][(sp + lg) * 12];
#pragma unroll
    for (int q = 0; q < 4; ++q) {
      const int mk = 8 >> q;
      float z = 0.f, xx = 0.f, yy = 0.f;
#pragma unroll
      for (int i = 0; i < 16; ++i) z += (i & mk) ? -n2[i] : n2[i];
#pragma unroll
      for (int i = 0; i < 16; ++i)
        if (!(i & mk)) {
          const int j = i | mk;
          xx += PRE(i) * PRE(j) + PIM(i) * PIM(j);
          yy += PRE(i) * PIM(j) - PIM(i) * PRE(j);
        }
      qrow[q]     = z;
      qrow[4 + q] = 2.f * xx;
      qrow[8 + q] = 2.f * yy;
    }
#undef PRE
#undef PIM
  }
  __syncthreads();

  // ---- LayerNorm over 192 features (two lanes per token) ----
  {
    const int f0 = lg * 96;
    float sum = 0.f, sum2 = 0.f;
    for (int f = 0; f < 96; ++f) {
      float v = q_lds[w][m][f0 + f];
      sum += v; sum2 += v * v;
    }
    sum  += __shfl_xor(sum, 16);
    sum2 += __shfl_xor(sum2, 16);
    float mu   = sum * (1.f / 192.f);
    float var  = sum2 * (1.f / 192.f) - mu * mu;
    float rstd = rsqrtf(var + 1e-5f);
    for (int f = 0; f < 96; ++f) {
      int fg = f0 + f;
      q_lds[w][m][fg] = (q_lds[w][m][fg] - mu) * rstd * ln_g[fg] + ln_b[fg];
    }
  }
  __syncthreads();

  // ---- GEMM1: h(16x192) @ W1^T(192x64), f16 WMMA ----
  v8f acc[4] = {};
#pragma unroll
  for (int kc = 0; kc < 6; ++kc) {
    v16h a;
#pragma unroll
    for (int h = 0; h < 16; ++h) {
      int j = h >> 1, rr = h & 1;
      int kk = kc * 32 + 16 * lg + 2 * j + rr;
      a[h] = (_Float16)q_lds[w][m][kk];
    }
#pragma unroll
    for (int nt = 0; nt < 4; ++nt) {
      int n = nt * 16 + m;
      v16h b = *(const v16h*)(W1h + n * 192 + kc * 32 + lg * 16);
      acc[nt] = __builtin_amdgcn_wmma_f32_16x16x32_f16(
          false, a, false, b, (short)0, acc[nt], false, false);
    }
  }
  __syncthreads();

  // ---- bias + Swish, restage (reuse q_lds region, stride 66) ----
  float* h1 = &q_lds[w][0][0];
#pragma unroll
  for (int nt = 0; nt < 4; ++nt) {
    int n = nt * 16 + m;
    float bb = b1[n];
#pragma unroll
    for (int j = 0; j < 8; ++j) {
      float v  = acc[nt][j] + bb;
      float sw = v / (1.f + __expf(-v));
      h1[(j + 8 * lg) * 66 + n] = sw;
    }
  }
  __syncthreads();

  // ---- GEMM2: h1(16x64) @ W2^T(64x64), f16 WMMA ----
  v8f acc2[4] = {};
#pragma unroll
  for (int kc = 0; kc < 2; ++kc) {
    v16h a;
#pragma unroll
    for (int h = 0; h < 16; ++h) {
      int j = h >> 1, rr = h & 1;
      int kk = kc * 32 + 16 * lg + 2 * j + rr;
      a[h] = (_Float16)h1[m * 66 + kk];
    }
#pragma unroll
    for (int nt = 0; nt < 4; ++nt) {
      int n = nt * 16 + m;
      v16h b = *(const v16h*)(W2h + n * 64 + kc * 32 + lg * 16);
      acc2[nt] = __builtin_amdgcn_wmma_f32_16x16x32_f16(
          false, a, false, b, (short)0, acc2[nt], false, false);
    }
  }

  // ---- epilogue: +b2, store ----
#pragma unroll
  for (int nt = 0; nt < 4; ++nt) {
    int n = nt * 16 + m;
    float bb = b2[n];
#pragma unroll
    for (int j = 0; j < 8; ++j) {
      out[(tok0 + j + 8 * lg) * 64 + n] = acc2[nt][j] + bb;
    }
  }
}

// ---------------------------------------------------------------------------
// Inputs (setup_inputs order): 0:x 1:qweights 2:ln_g 3:ln_b 4:W1 5:b1 6:W2 7:b2
// Workspace: [0,32KB) Are/Aim fp32; [32KB,+24KB) W1h f16; then W2h f16 (8KB).
// ---------------------------------------------------------------------------
extern "C" void kernel_launch(void* const* d_in, const int* in_sizes, int n_in,
                              void* d_out, int out_size, void* d_ws, size_t ws_size,
                              hipStream_t stream) {
  const float* x    = (const float*)d_in[0];
  const float* qw   = (const float*)d_in[1];
  const float* ln_g = (const float*)d_in[2];
  const float* ln_b = (const float*)d_in[3];
  const float* W1   = (const float*)d_in[4];
  const float* b1   = (const float*)d_in[5];
  const float* W2   = (const float*)d_in[6];
  const float* b2   = (const float*)d_in[7];
  float* out = (float*)d_out;

  float*    A   = (float*)d_ws;                         // 32 KB
  _Float16* W1h = (_Float16*)((char*)d_ws + 32768);     // 24 KB
  _Float16* W2h = W1h + 64 * 192;                       //  8 KB

  build_unitaries<<<16, 16, 0, stream>>>(qw, A);
  convert_weights<<<64, 256, 0, stream>>>(W1, W2, W1h, W2h);
  fused_forward<<<TOKENS / 32, 64, 0, stream>>>(x, A, ln_g, ln_b, W1h, b1, W2h, b2, out);
}
